// HRNetW30classifier_30176440222404
// MI455X (gfx1250) — matrix operand order repaired and checked by reading, not compiled, not run on hardware
//
#include <hip/hip_runtime.h>

// ---------------------------------------------------------------------------
// Fused 1000-head linear classifier == GEMM (8192x2048) x (2048x1000) + bias
// CDNA5 / gfx1250, wave32. bf16x3 split-precision WMMA (near-fp32 accuracy).
// Phase 1: fp32 -> bf16 hi/lo planes in workspace (memory-bound, L2-resident).
// Phase 2: WMMA GEMM, double-buffered LDS, async global->LDS staging
//          (GLOBAL_LOAD_ASYNC_TO_LDS_B128 / ASYNCcnt) when available.
// Falls back to a single-pass fused kernel if the workspace is too small.
// ---------------------------------------------------------------------------

typedef __attribute__((ext_vector_type(16))) __bf16        v16bf;
typedef __attribute__((ext_vector_type(8)))  float         v8f;
typedef __attribute__((ext_vector_type(8)))  unsigned int  v8u;
typedef __attribute__((ext_vector_type(4)))  int           v4i;

#define BATCH    8192
#define FEAT     2048
#define NCLS     1000
#define NCLS_PAD 1024

// Workspace layout (bf16 elements)
#define XHI_OFF  ((size_t)0)
#define XLO_OFF  ((size_t)BATCH * FEAT)
#define WHI_OFF  ((size_t)2 * BATCH * FEAT)
#define WLO_OFF  ((size_t)2 * BATCH * FEAT + (size_t)NCLS_PAD * FEAT)
#define WS_ELEMS ((size_t)2 * BATCH * FEAT + (size_t)2 * NCLS_PAD * FEAT)
#define WS_BYTES (WS_ELEMS * 2)

union FragBF {
  v8u   u;   // 8 dwords = 16 bf16
  v16bf bf;
};

__device__ __forceinline__ unsigned short f32_bf16_rne(float f) {
  unsigned int u = __builtin_bit_cast(unsigned int, f);
  u += 0x7fffu + ((u >> 16) & 1u);          // round-to-nearest-even
  return (unsigned short)(u >> 16);
}
__device__ __forceinline__ float bf16_f32(unsigned short h) {
  unsigned int u = ((unsigned int)h) << 16;
  return __builtin_bit_cast(float, u);
}

// ---- CDNA5 async global->LDS copy (compile-time gated) --------------------
#if defined(__has_builtin)
#if __has_builtin(__builtin_amdgcn_global_load_async_to_lds_b128)
#define HAVE_ASYNC_LDS 1
#endif
#endif
#ifndef HAVE_ASYNC_LDS
#define HAVE_ASYNC_LDS 0
#endif

#if HAVE_ASYNC_LDS
// Prototype (from clang diagnostic): (int4 AS1*, int4 AS3*, imm int, imm int)
typedef __attribute__((address_space(1))) v4i g_v4i;
typedef __attribute__((address_space(3))) v4i l_v4i;

__device__ __forceinline__ void async_cp16(const void* g, void* l) {
  // Generic LDS pointer: low 32 bits are the LDS byte offset.
  __builtin_amdgcn_global_load_async_to_lds_b128(
      (g_v4i*)(unsigned long long)g,
      (l_v4i*)(unsigned int)(unsigned long long)l,
      0, 0);
}
__device__ __forceinline__ void async_fence() {
  asm volatile("s_wait_asynccnt 0x0" ::: "memory");
}
#else
__device__ __forceinline__ void async_fence() {}
#endif

// ---------------------------------------------------------------------------
// Pre-pass: split fp32 into bf16 hi/lo planes. Memory-bound; VALU is free here.
// ---------------------------------------------------------------------------
__global__ __launch_bounds__(256)
void split_x_kernel(const float* __restrict__ X,
                    unsigned short* __restrict__ hi,
                    unsigned short* __restrict__ lo) {
  const size_t q = (size_t)blockIdx.x * 256 + threadIdx.x;  // float4 index
  const float4 v = ((const float4*)X)[q];
  const float vals[4] = {v.x, v.y, v.z, v.w};
  unsigned short h[4], l[4];
#pragma unroll
  for (int e = 0; e < 4; ++e) {
    h[e] = f32_bf16_rne(vals[e]);
    l[e] = f32_bf16_rne(vals[e] - bf16_f32(h[e]));
  }
  uint2 hp, lp;
  hp.x = (unsigned)h[0] | ((unsigned)h[1] << 16);
  hp.y = (unsigned)h[2] | ((unsigned)h[3] << 16);
  lp.x = (unsigned)l[0] | ((unsigned)l[1] << 16);
  lp.y = (unsigned)l[2] | ((unsigned)l[3] << 16);
  *(uint2*)(hi + q * 4) = hp;
  *(uint2*)(lo + q * 4) = lp;
}

__global__ __launch_bounds__(256)
void split_w_kernel(const float* __restrict__ W,
                    unsigned short* __restrict__ hi,
                    unsigned short* __restrict__ lo) {
  const size_t q  = (size_t)blockIdx.x * 256 + threadIdx.x;  // float4 idx (padded)
  const int   row = (int)(q / (FEAT / 4));                   // padded class row
  float4 v = make_float4(0.f, 0.f, 0.f, 0.f);
  if (row < NCLS) v = ((const float4*)W)[q];                 // rows coincide
  const float vals[4] = {v.x, v.y, v.z, v.w};
  unsigned short h[4], l[4];
#pragma unroll
  for (int e = 0; e < 4; ++e) {
    h[e] = f32_bf16_rne(vals[e]);
    l[e] = f32_bf16_rne(vals[e] - bf16_f32(h[e]));
  }
  uint2 hp, lp;
  hp.x = (unsigned)h[0] | ((unsigned)h[1] << 16);
  hp.y = (unsigned)h[2] | ((unsigned)h[3] << 16);
  lp.x = (unsigned)l[0] | ((unsigned)l[1] << 16);
  lp.y = (unsigned)l[2] | ((unsigned)l[3] << 16);
  *(uint2*)(hi + q * 4) = hp;
  *(uint2*)(lo + q * 4) = lp;
}

// ---------------------------------------------------------------------------
// Main GEMM: bf16 planes -> (async) LDS -> WMMA. Block tile 128x128,
// 8 wave32s as 2(M) x 4(N); wave tile 64x32 = 4x2 accumulators.
// Double-buffered LDS: one barrier per K-step, stage(i+1) overlaps compute(i).
// ---------------------------------------------------------------------------
#define M_BLK2  128
#define N_BLK2  128
#define K_STEP  32
#define LDS_S2  40   // padded row stride (elements); 80 B keeps 16 B alignment

__device__ __forceinline__ void stage_tiles(
    const unsigned short* __restrict__ Xhi, const unsigned short* __restrict__ Xlo,
    const unsigned short* __restrict__ Whi, const unsigned short* __restrict__ Wlo,
    unsigned short (*dA)[M_BLK2][LDS_S2],     // [plane][row][k] of chosen buffer
    unsigned short (*dB)[N_BLK2][LDS_S2],
    int m_blk0, int n_blk0, int k0, int tid) {
#pragma unroll
  for (int i = 0; i < 2; ++i) {
    const int q   = tid + i * 256;
    const int row = q >> 2;                   // 4 x 8-element chunks per row
    const int c8  = (q & 3) << 3;
    const size_t ax = (size_t)(m_blk0 + row) * FEAT + k0 + c8;
    const size_t bx = (size_t)(n_blk0 + row) * FEAT + k0 + c8;
#if HAVE_ASYNC_LDS
    async_cp16(Xhi + ax, &dA[0][row][c8]);
    async_cp16(Xlo + ax, &dA[1][row][c8]);
    async_cp16(Whi + bx, &dB[0][row][c8]);
    async_cp16(Wlo + bx, &dB[1][row][c8]);
#else
    const uint4 ah = *(const uint4*)(Xhi + ax);
    const uint4 al = *(const uint4*)(Xlo + ax);
    const uint4 bh = *(const uint4*)(Whi + bx);
    const uint4 bl = *(const uint4*)(Wlo + bx);
    *(uint4*)&dA[0][row][c8] = ah;
    *(uint4*)&dA[1][row][c8] = al;
    *(uint4*)&dB[0][row][c8] = bh;
    *(uint4*)&dB[1][row][c8] = bl;
#endif
  }
}

__global__ __launch_bounds__(256)
void gemm_planes_kernel(const unsigned short* __restrict__ Xhi,
                        const unsigned short* __restrict__ Xlo,
                        const unsigned short* __restrict__ Whi,
                        const unsigned short* __restrict__ Wlo,
                        const float* __restrict__ Bias,
                        float* __restrict__ Out) {
  __shared__ unsigned short sA[2][2][M_BLK2][LDS_S2];  // [buf][plane][row][k]
  __shared__ unsigned short sB[2][2][N_BLK2][LDS_S2];

  const int tid   = threadIdx.x;
  const int lane  = tid & 31;
  const int wave  = tid >> 5;
  const int wm    = wave >> 2;     // 0..1
  const int wn    = wave & 3;      // 0..3
  const int lrow  = lane & 15;
  const int lhalf = lane >> 4;

  const int m_blk0 = blockIdx.x * M_BLK2;
  const int n_blk0 = blockIdx.y * N_BLK2;   // padded W: no load guards needed

  v8f acc[4][2];
#pragma unroll
  for (int mt = 0; mt < 4; ++mt)
#pragma unroll
    for (int nt = 0; nt < 2; ++nt)
      acc[mt][nt] = (v8f){0.f, 0.f, 0.f, 0.f, 0.f, 0.f, 0.f, 0.f};

  // Prologue: stage first tile into buffer 0.
  stage_tiles(Xhi, Xlo, Whi, Wlo, sA[0], sB[0], m_blk0, n_blk0, 0, tid);
  async_fence();
  __syncthreads();

  for (int k0 = 0; k0 < FEAT; k0 += K_STEP) {
    const int cur = (k0 >> 5) & 1;
    const int nxt = cur ^ 1;

    // Stage next tile into the other buffer (overlaps compute below).
    if (k0 + K_STEP < FEAT)
      stage_tiles(Xhi, Xlo, Whi, Wlo, sA[nxt], sB[nxt], m_blk0, n_blk0,
                  k0 + K_STEP, tid);

    // Fragment gather per ISA 7.12.2:
    // A: lane = 16*half + m ; dword t -> K = 2t + 8*half + 8*(t>=4)
    FragBF a[4][2];
#pragma unroll
    for (int mt = 0; mt < 4; ++mt) {
      const int r = wm * 64 + mt * 16 + lrow;
#pragma unroll
      for (int t = 0; t < 8; ++t) {
        const int kb = 2 * t + 8 * lhalf + ((t >= 4) ? 8 : 0);
        a[mt][0].u[t] = *(const unsigned int*)&sA[cur][0][r][kb];
        a[mt][1].u[t] = *(const unsigned int*)&sA[cur][1][r][kb];
      }
    }
    // B: lane = 16*half + n ; dword t -> K = 2t + 8*half + 8*(t>=4)
    FragBF b[2][2];
#pragma unroll
    for (int nt = 0; nt < 2; ++nt) {
      const int c = wn * 32 + nt * 16 + lrow;
#pragma unroll
      for (int t = 0; t < 8; ++t) {
        const int kb = 2 * t + 8 * lhalf + ((t >= 4) ? 8 : 0);
        b[nt][0].u[t] = *(const unsigned int*)&sB[cur][0][c][kb];
        b[nt][1].u[t] = *(const unsigned int*)&sB[cur][1][c][kb];
      }
    }

    // bf16x3: hi*hi + lo*hi + hi*lo -> 24 WMMAs / K-step / wave
#pragma unroll
    for (int mt = 0; mt < 4; ++mt) {
#pragma unroll
      for (int nt = 0; nt < 2; ++nt) {
        acc[mt][nt] = __builtin_amdgcn_wmma_f32_16x16x32_bf16(
            false, a[mt][0].bf, false, b[nt][0].bf, (short)0, acc[mt][nt],
            false, false);
        acc[mt][nt] = __builtin_amdgcn_wmma_f32_16x16x32_bf16(
            false, a[mt][1].bf, false, b[nt][0].bf, (short)0, acc[mt][nt],
            false, false);
        acc[mt][nt] = __builtin_amdgcn_wmma_f32_16x16x32_bf16(
            false, a[mt][0].bf, false, b[nt][1].bf, (short)0, acc[mt][nt],
            false, false);
      }
    }

    // Next buffer staged + this buffer's reads done -> single barrier/iter.
    async_fence();
    __syncthreads();
  }

  // Epilogue. D layout: col = lane&15, row = r + 8*half.
#pragma unroll
  for (int nt = 0; nt < 2; ++nt) {
    const int col = n_blk0 + wn * 32 + nt * 16 + lrow;
    if (col < NCLS) {
      const float bias = Bias[col];
#pragma unroll
      for (int mt = 0; mt < 4; ++mt) {
        const int row0 = m_blk0 + wm * 64 + mt * 16 + 8 * lhalf;
#pragma unroll
        for (int r = 0; r < 8; ++r)
          Out[(size_t)(row0 + r) * NCLS + col] = acc[mt][nt][r] + bias;
      }
    }
  }
}

// ---------------------------------------------------------------------------
// Fallback: single-pass fused kernel (conversion inside loop) — used only if
// workspace is too small. Block tile 128x64, wave tile 32x32.
// ---------------------------------------------------------------------------
#define M_BLK1  128
#define N_BLK1  64
#define LDS_S1  36

__global__ __launch_bounds__(256)
void gemm_fused_kernel(const float* __restrict__ X, const float* __restrict__ W,
                       const float* __restrict__ Bias, float* __restrict__ Out) {
  __shared__ unsigned short sAhi[M_BLK1][LDS_S1];
  __shared__ unsigned short sAlo[M_BLK1][LDS_S1];
  __shared__ unsigned short sBhi[N_BLK1][LDS_S1];
  __shared__ unsigned short sBlo[N_BLK1][LDS_S1];

  const int tid = threadIdx.x, lane = tid & 31, wave = tid >> 5;
  const int wm = wave >> 1, wn = wave & 1;
  const int lrow = lane & 15, lhalf = lane >> 4;
  const int m_blk0 = blockIdx.x * M_BLK1;
  const int n_blk0 = blockIdx.y * N_BLK1;

  v8f acc[2][2];
#pragma unroll
  for (int mt = 0; mt < 2; ++mt)
#pragma unroll
    for (int nt = 0; nt < 2; ++nt)
      acc[mt][nt] = (v8f){0.f, 0.f, 0.f, 0.f, 0.f, 0.f, 0.f, 0.f};

  for (int k0 = 0; k0 < FEAT; k0 += K_STEP) {
    __syncthreads();
#pragma unroll
    for (int i = 0; i < 4; ++i) {
      const int f = tid + i * 256, r = f >> 3, c4 = (f & 7) << 2;
      const float4 v = *(const float4*)(X + (size_t)(m_blk0 + r) * FEAT + k0 + c4);
      const float vals[4] = {v.x, v.y, v.z, v.w};
#pragma unroll
      for (int e = 0; e < 4; ++e) {
        const unsigned short hi = f32_bf16_rne(vals[e]);
        sAhi[r][c4 + e] = hi;
        sAlo[r][c4 + e] = f32_bf16_rne(vals[e] - bf16_f32(hi));
      }
    }
#pragma unroll
    for (int i = 0; i < 2; ++i) {
      const int f = tid + i * 256, r = f >> 3, c4 = (f & 7) << 2;
      const int cls = n_blk0 + r;
      float4 v = make_float4(0.f, 0.f, 0.f, 0.f);
      if (cls < NCLS) v = *(const float4*)(W + (size_t)cls * FEAT + k0 + c4);
      const float vals[4] = {v.x, v.y, v.z, v.w};
#pragma unroll
      for (int e = 0; e < 4; ++e) {
        const unsigned short hi = f32_bf16_rne(vals[e]);
        sBhi[r][c4 + e] = hi;
        sBlo[r][c4 + e] = f32_bf16_rne(vals[e] - bf16_f32(hi));
      }
    }
    __syncthreads();

    FragBF a_hi[2], a_lo[2], b_hi[2], b_lo[2];
#pragma unroll
    for (int mt = 0; mt < 2; ++mt) {
      const int r = wm * 32 + mt * 16 + lrow;
#pragma unroll
      for (int t = 0; t < 8; ++t) {
        const int kb = 2 * t + 8 * lhalf + ((t >= 4) ? 8 : 0);
        a_hi[mt].u[t] = *(const unsigned int*)&sAhi[r][kb];
        a_lo[mt].u[t] = *(const unsigned int*)&sAlo[r][kb];
      }
    }
#pragma unroll
    for (int nt = 0; nt < 2; ++nt) {
      const int c = wn * 32 + nt * 16 + lrow;
#pragma unroll
      for (int t = 0; t < 8; ++t) {
        const int kb = 2 * t + 8 * lhalf + ((t >= 4) ? 8 : 0);
        b_hi[nt].u[t] = *(const unsigned int*)&sBhi[c][kb];
        b_lo[nt].u[t] = *(const unsigned int*)&sBlo[c][kb];
      }
    }
#pragma unroll
    for (int mt = 0; mt < 2; ++mt)
#pragma unroll
      for (int nt = 0; nt < 2; ++nt) {
        acc[mt][nt] = __builtin_amdgcn_wmma_f32_16x16x32_bf16(
            false, a_hi[mt].bf, false, b_hi[nt].bf, (short)0, acc[mt][nt], false, false);
        acc[mt][nt] = __builtin_amdgcn_wmma_f32_16x16x32_bf16(
            false, a_lo[mt].bf, false, b_hi[nt].bf, (short)0, acc[mt][nt], false, false);
        acc[mt][nt] = __builtin_amdgcn_wmma_f32_16x16x32_bf16(
            false, a_hi[mt].bf, false, b_lo[nt].bf, (short)0, acc[mt][nt], false, false);
      }
  }

#pragma unroll
  for (int nt = 0; nt < 2; ++nt) {
    const int col = n_blk0 + wn * 32 + nt * 16 + lrow;
    if (col < NCLS) {
      const float bias = Bias[col];
#pragma unroll
      for (int mt = 0; mt < 2; ++mt) {
        const int row0 = m_blk0 + wm * 32 + mt * 16 + 8 * lhalf;
#pragma unroll
        for (int r = 0; r < 8; ++r)
          Out[(size_t)(row0 + r) * NCLS + col] = acc[mt][nt][r] + bias;
      }
    }
  }
}

extern "C" void kernel_launch(void* const* d_in, const int* in_sizes, int n_in,
                              void* d_out, int out_size, void* d_ws,
                              size_t ws_size, hipStream_t stream) {
  (void)in_sizes; (void)n_in; (void)out_size;
  const float* x    = (const float*)d_in[0];
  const float* W    = (const float*)d_in[1];
  const float* bias = (const float*)d_in[2];
  float* out        = (float*)d_out;

  if (ws_size >= WS_BYTES) {
    unsigned short* ws  = (unsigned short*)d_ws;
    unsigned short* xhi = ws + XHI_OFF;
    unsigned short* xlo = ws + XLO_OFF;
    unsigned short* whi = ws + WHI_OFF;
    unsigned short* wlo = ws + WLO_OFF;

    split_x_kernel<<<dim3((BATCH * FEAT / 4) / 256), dim3(256), 0, stream>>>(
        x, xhi, xlo);
    split_w_kernel<<<dim3((NCLS_PAD * FEAT / 4) / 256), dim3(256), 0, stream>>>(
        W, whi, wlo);
    gemm_planes_kernel<<<dim3(BATCH / M_BLK2, NCLS_PAD / N_BLK2), dim3(256), 0,
                         stream>>>(xhi, xlo, whi, wlo, bias, out);
  } else {
    gemm_fused_kernel<<<dim3(BATCH / M_BLK1, (NCLS + N_BLK1 - 1) / N_BLK1),
                        dim3(256), 0, stream>>>(x, W, bias, out);
  }
}
